// AttentionLayer_59528246722882
// MI455X (gfx1250) — compile-verified, matrix operand order checked by reading, not run on hardware
//
#include <hip/hip_runtime.h>

// ---------------------------------------------------------------------------
// Types for CDNA5 WMMA (wave32): v16bf A/B fragments, v8f accumulators.
// ---------------------------------------------------------------------------
typedef __attribute__((ext_vector_type(16))) __bf16 v16bf;
typedef __attribute__((ext_vector_type(8)))  __bf16 v8bf;
typedef __attribute__((ext_vector_type(8)))  float  v8f;

static __device__ __forceinline__ v8bf ld8(const __bf16* p) {
    return *(const v8bf*)p;   // 16-byte aligned by construction everywhere below
}
static __device__ __forceinline__ v16bf combine(v8bf lo, v8bf hi) {
    return __builtin_shufflevector(lo, hi, 0,1,2,3,4,5,6,7,8,9,10,11,12,13,14,15);
}
static __device__ __forceinline__ v8f wmma_bf16(v16bf a, v16bf b, v8f c) {
    // D(f32 16x16) = A(16x32 bf16) * B(32x16 bf16) + C
    return __builtin_amdgcn_wmma_f32_16x16x32_bf16(false, a, false, b, (short)0, c,
                                                   false, false);
}

// CDNA5 async copy: global -> LDS, 16 bytes per lane, tracked by ASYNCcnt.
static __device__ __forceinline__ void async_ld16(unsigned lds_addr, const void* g) {
    asm volatile("global_load_async_to_lds_b128 %0, %1, off"
                 :: "v"(lds_addr), "v"((unsigned long long)(uintptr_t)g)
                 : "memory");
}
static __device__ __forceinline__ void wait_async0() {
    asm volatile("s_wait_asynccnt 0x0" ::: "memory");
}

// Problem constants
#define BB 16
#define SS 1024
#define DD 1024
#define HH 16
#define DH 64

// LDS staging geometry: 128 rows x 32 bf16 per tile, padded row stride 80B
// (20 banks per row step, gcd(20,64)=4 -> 16 lanes hit 16 distinct banks).
#define ROWB  80
#define TILEB (128 * ROWB)   // 10240 B per tile

// ---------------------------------------------------------------------------
// f32 -> bf16 elementwise convert
// ---------------------------------------------------------------------------
__global__ __launch_bounds__(256) void cvt_kernel(const float* __restrict__ in,
                                                  __bf16* __restrict__ out, int n) {
    int i = blockIdx.x * 256 + threadIdx.x;
    if (i < n) out[i] = (__bf16)in[i];
}

// ---------------------------------------------------------------------------
// GEMM:  C[m,n] = sum_k A[m,k] * W[n,k] + bias[n]        (i.e.  A @ W^T + b)
// A: bf16 [16384,1024] row-major, W: bf16 [1024,1024] row-major.
// Block 256 thr = 8 waves (2 m x 4 n); wave tile 64x32; block tile 128x128.
// Block tiles of A and B are staged in LDS with async-to-LDS copies, double
// buffered: per k-step, 4 async 16B issues/thread overlap 8 WMMAs + 12
// ds_load_b128 fragment reads.
// mode 0: store bf16 row-major, value*scale        (Q with 1/64, K with 1)
// mode 1: store bf16 transposed per head -> Vt[B,H,DH,S]
// mode 2: store f32 row-major                      (output projection)
// ---------------------------------------------------------------------------
__global__ __launch_bounds__(256) void gemm_kernel(const __bf16* __restrict__ A,
                                                   const __bf16* __restrict__ W,
                                                   const float* __restrict__ bias,
                                                   void* __restrict__ out,
                                                   float scale, int mode) {
    __shared__ __align__(16) char lds[2][2][TILEB];   // [buf][A=0/B=1][tile]

    const int lane = threadIdx.x & 31;
    const int wid  = threadIdx.x >> 5;
    const int half = lane >> 4;
    const int l16  = lane & 15;
    const int wr = wid & 1, wc = wid >> 1;
    const int mblk = blockIdx.x * 128;
    const int nblk = blockIdx.y * 128;

    const unsigned lbase = (unsigned)(uintptr_t)&lds[0][0][0];

    // Per-thread copy assignment: 512 16B chunks per tile, 2 per thread.
    const int cid0 = threadIdx.x * 2;

    // Issue the async stage of one k-slab into buffer `buf`.
    auto issue = [&](int buf, int k0) {
#pragma unroll
        for (int j = 0; j < 2; ++j) {
            const int cid = cid0 + j;          // 0..511
            const int row = cid >> 2;          // 0..127
            const int ch  = cid & 3;           // 16B chunk within 64B row
            const unsigned loff = (unsigned)(row * ROWB + ch * 16);
            async_ld16(lbase + (unsigned)(buf * 2 * TILEB) + loff,
                       A + (size_t)(mblk + row) * DD + k0 + ch * 8);
            async_ld16(lbase + (unsigned)((buf * 2 + 1) * TILEB) + loff,
                       W + (size_t)(nblk + row) * DD + k0 + ch * 8);
        }
    };

    v8f acc[4][2];
#pragma unroll
    for (int tm = 0; tm < 4; ++tm)
#pragma unroll
        for (int tn = 0; tn < 2; ++tn) acc[tm][tn] = (v8f)0.0f;

    issue(0, 0);
    wait_async0();
    __syncthreads();

    for (int k0 = 0; k0 < DD; k0 += 32) {
        const int cur = (k0 >> 5) & 1;
        const bool more = (k0 + 32) < DD;
        if (more) issue(cur ^ 1, k0 + 32);

        const char* aT = &lds[cur][0][0];
        const char* bT = &lds[cur][1][0];

        v16bf a[4], b[2];
#pragma unroll
        for (int tm = 0; tm < 4; ++tm) {
            const __bf16* p =
                (const __bf16*)(aT + (wr * 64 + tm * 16 + l16) * ROWB);
            a[tm] = combine(ld8(p + half * 8), ld8(p + 16 + half * 8));
        }
#pragma unroll
        for (int tn = 0; tn < 2; ++tn) {
            const __bf16* p =
                (const __bf16*)(bT + (wc * 32 + tn * 16 + l16) * ROWB);
            b[tn] = combine(ld8(p + half * 8), ld8(p + 16 + half * 8));
        }
#pragma unroll
        for (int tm = 0; tm < 4; ++tm)
#pragma unroll
            for (int tn = 0; tn < 2; ++tn)
                acc[tm][tn] = wmma_bf16(a[tm], b[tn], acc[tm][tn]);

        if (more) {
            wait_async0();
            __syncthreads();
        }
    }

    // Epilogue.  C layout: lane l holds col n = l16, rows m = r + half*8.
    const int mbase = mblk + wr * 64;
    const int nbase = nblk + wc * 32;
#pragma unroll
    for (int tm = 0; tm < 4; ++tm) {
#pragma unroll
        for (int tn = 0; tn < 2; ++tn) {
            const int colg = nbase + tn * 16 + l16;
            const float bv = bias[colg];
            const int rowb = mbase + tm * 16 + half * 8;
            v8f c = acc[tm][tn];
            if (mode == 0) {
                __bf16* o = (__bf16*)out;
#pragma unroll
                for (int r = 0; r < 8; ++r)
                    o[(size_t)(rowb + r) * DD + colg] = (__bf16)((c[r] + bv) * scale);
            } else if (mode == 1) {
                // Vt[b][h][dh][s], 8 consecutive s values -> one 16B store
                const int h = colg >> 6, dh = colg & 63;
                const int bidx = rowb >> 10, s = rowb & (SS - 1);
                __bf16* o = (__bf16*)out +
                            (((size_t)(bidx * HH + h) * DH + dh) << 10) + s;
                v8bf pk;
#pragma unroll
                for (int r = 0; r < 8; ++r) pk[r] = (__bf16)(c[r] + bv);
                *(v8bf*)o = pk;
            } else {
                float* o = (float*)out;
#pragma unroll
                for (int r = 0; r < 8; ++r)
                    o[(size_t)(rowb + r) * DD + colg] = c[r] + bv;
            }
        }
    }
}

// ---------------------------------------------------------------------------
// Flash-style attention. Each wave owns 16 queries of one (b,h); block = 8
// waves = 128 queries. Scores computed TRANSPOSED (M=keys, N=queries) so the
// softmax reduction is over accumulator VGPRs + one half-wave shuffle, and the
// resulting P tile is exactly the B-fragment layout for the ctx^T = Vt * P
// WMMA.  Q is pre-scaled by 1/64; mask value -1e8 matches the reference.
// ---------------------------------------------------------------------------
__global__ __launch_bounds__(256) void attn_kernel(const __bf16* __restrict__ Q,
                                                   const __bf16* __restrict__ Km,
                                                   const __bf16* __restrict__ Vt,
                                                   const int* __restrict__ padMask,
                                                   __bf16* __restrict__ ctx) {
    const int lane = threadIdx.x & 31;
    const int wid  = threadIdx.x >> 5;
    const int half = lane >> 4;
    const int l16  = lane & 15;

    const int bh   = blockIdx.x >> 3;       // 0..255
    const int qblk = blockIdx.x & 7;
    const int b = bh >> 4, h = bh & 15;
    const int q = qblk * 128 + wid * 16 + l16;

    // Q fragment (B-matrix, d-major): lane l holds query q=l16
    const __bf16* qrow = Q + (size_t)(b * SS + q) * DD + h * DH;
    const v16bf qf0 = combine(ld8(qrow + half * 8),      ld8(qrow + 16 + half * 8));
    const v16bf qf1 = combine(ld8(qrow + 32 + half * 8), ld8(qrow + 48 + half * 8));
    const bool qpad = padMask[b * SS + q] != 0;

    const int*    pm  = padMask + b * SS;
    const __bf16* kb  = Km + (size_t)(b * SS) * DD + h * DH;
    const __bf16* vtb = Vt + ((size_t)(b * HH + h) * DH) * SS;

    float m = -3.0e38f, lsum = 0.0f;
    v8f cacc[4];
#pragma unroll
    for (int t = 0; t < 4; ++t) cacc[t] = (v8f)0.0f;

    for (int kc = 0; kc < SS; kc += 32) {
        // ---- scores^T for 32 keys x 16 queries (2 WMMA tiles, K over DH) ----
        v8f s0 = (v8f)0.0f, s1 = (v8f)0.0f;
        {
            const __bf16* kr = kb + (size_t)(kc + l16) * DD;
            v16bf a0 = combine(ld8(kr + half * 8),      ld8(kr + 16 + half * 8));
            v16bf a1 = combine(ld8(kr + 32 + half * 8), ld8(kr + 48 + half * 8));
            s0 = wmma_bf16(a0, qf0, s0);
            s0 = wmma_bf16(a1, qf1, s0);
            kr += (size_t)16 * DD;
            a0 = combine(ld8(kr + half * 8),      ld8(kr + 16 + half * 8));
            a1 = combine(ld8(kr + 32 + half * 8), ld8(kr + 48 + half * 8));
            s1 = wmma_bf16(a0, qf0, s1);
            s1 = wmma_bf16(a1, qf1, s1);
        }

        // ---- key pad bits for the 16 keys this lane holds ----
        int kpad[16];
        {
            int4 p0 = *(const int4*)(pm + kc + half * 8);
            int4 p1 = *(const int4*)(pm + kc + half * 8 + 4);
            int4 p2 = *(const int4*)(pm + kc + 16 + half * 8);
            int4 p3 = *(const int4*)(pm + kc + 16 + half * 8 + 4);
            kpad[0]=p0.x; kpad[1]=p0.y; kpad[2]=p0.z; kpad[3]=p0.w;
            kpad[4]=p1.x; kpad[5]=p1.y; kpad[6]=p1.z; kpad[7]=p1.w;
            kpad[8]=p2.x; kpad[9]=p2.y; kpad[10]=p2.z; kpad[11]=p2.w;
            kpad[12]=p3.x; kpad[13]=p3.y; kpad[14]=p3.z; kpad[15]=p3.w;
        }

        // ---- mask + online softmax (per-query stats live per lane) ----
        float p[16];
        float cmax = -3.0e38f;
#pragma unroll
        for (int i = 0; i < 16; ++i) {
            float v = (i < 8) ? s0[i] : s1[i - 8];
            v = (qpad || (kpad[i] != 0)) ? -1.0e8f : v;
            p[i] = v;
            cmax = fmaxf(cmax, v);
        }
        cmax = fmaxf(cmax, __shfl_xor(cmax, 16));
        const float mnew = fmaxf(m, cmax);
        const float ef = __expf(m - mnew);
        float csum = 0.0f;
#pragma unroll
        for (int i = 0; i < 16; ++i) {
            p[i] = __expf(p[i] - mnew);
            csum += p[i];
        }
        csum += __shfl_xor(csum, 16);
        lsum = lsum * ef + csum;
        m = mnew;
#pragma unroll
        for (int t = 0; t < 4; ++t)
#pragma unroll
            for (int r = 0; r < 8; ++r) cacc[t][r] *= ef;

        // ---- P tile already sits in B-fragment layout: pack to bf16 ----
        v16bf pf;
#pragma unroll
        for (int i = 0; i < 16; ++i) pf[i] = (__bf16)p[i];

        // ---- ctx^T += Vt(16d x 32k) * P(32k x 16q), 4 tiles over DH ----
#pragma unroll
        for (int t = 0; t < 4; ++t) {
            const __bf16* vr = vtb + (size_t)(t * 16 + l16) * SS + kc;
            v16bf vf = combine(ld8(vr + half * 8), ld8(vr + 16 + half * 8));
            cacc[t] = wmma_bf16(vf, pf, cacc[t]);
        }
    }

    // ---- normalize and store ctx (row-major [B*S, D], bf16) ----
    const float inv = 1.0f / lsum;
    __bf16* crow = ctx + (size_t)(b * SS + q) * DD + h * DH;
#pragma unroll
    for (int t = 0; t < 4; ++t)
#pragma unroll
        for (int r = 0; r < 8; ++r)
            crow[t * 16 + half * 8 + r] = (__bf16)(cacc[t][r] * inv);
}

// ---------------------------------------------------------------------------
// Residual + LayerNorm (Bessel-corrected std, eps added to std).
// One 256-thread block per row of 1024.
// ---------------------------------------------------------------------------
__global__ __launch_bounds__(256) void ln_kernel(const float* __restrict__ enc,
                                                 const float* __restrict__ att,
                                                 const float* __restrict__ gamma,
                                                 const float* __restrict__ beta,
                                                 float* __restrict__ out) {
    const int row = blockIdx.x;
    const float* e = enc + (size_t)row * DD;
    const float* a = att + (size_t)row * DD;
    float v[4], s = 0.0f, ss = 0.0f;
#pragma unroll
    for (int i = 0; i < 4; ++i) {
        const int c = threadIdx.x + i * 256;
        v[i] = e[c] + a[c];
        s += v[i];
        ss += v[i] * v[i];
    }
    for (int off = 16; off > 0; off >>= 1) {
        s  += __shfl_down(s, off);
        ss += __shfl_down(ss, off);
    }
    __shared__ float ws[8], wss[8];
    const int lane = threadIdx.x & 31, wid = threadIdx.x >> 5;
    if (lane == 0) { ws[wid] = s; wss[wid] = ss; }
    __syncthreads();
    float S = 0.0f, SQ = 0.0f;
#pragma unroll
    for (int w = 0; w < 8; ++w) { S += ws[w]; SQ += wss[w]; }
    const float mean = S * (1.0f / DD);
    const float var  = (SQ - (float)DD * mean * mean) * (1.0f / (DD - 1));
    const float inv  = 1.0f / (sqrtf(fmaxf(var, 0.0f)) + 1e-8f);
    float* o = out + (size_t)row * DD;
#pragma unroll
    for (int i = 0; i < 4; ++i) {
        const int c = threadIdx.x + i * 256;
        o[c] = gamma[c] * (v[i] - mean) * inv + beta[c];
    }
}

// ---------------------------------------------------------------------------
// Host-side orchestration
// ---------------------------------------------------------------------------
extern "C" void kernel_launch(void* const* d_in, const int* in_sizes, int n_in,
                              void* d_out, int out_size, void* d_ws, size_t ws_size,
                              hipStream_t stream) {
    const float* enc  = (const float*)d_in[0];
    const int*   pad  = (const int*)d_in[1];
    const float* Wq   = (const float*)d_in[2];
    const float* bq   = (const float*)d_in[3];
    const float* Wk   = (const float*)d_in[4];
    const float* bk   = (const float*)d_in[5];
    const float* Wv   = (const float*)d_in[6];
    const float* bv   = (const float*)d_in[7];
    const float* Wo   = (const float*)d_in[8];
    const float* bo   = (const float*)d_in[9];
    const float* gamma = (const float*)d_in[10];
    const float* beta  = (const float*)d_in[11];

    const size_t MBy = (size_t)1 << 20;
    char* w = (char*)d_ws;
    __bf16* Xb   = (__bf16*)(w + 0);          // 32 MB  [16384,1024] bf16
    __bf16* Wqb  = (__bf16*)(w + 32 * MBy);   //  2 MB
    __bf16* Wkb  = (__bf16*)(w + 34 * MBy);   //  2 MB
    __bf16* Wvb  = (__bf16*)(w + 36 * MBy);   //  2 MB
    __bf16* Wob  = (__bf16*)(w + 38 * MBy);   //  2 MB
    __bf16* Qb   = (__bf16*)(w + 40 * MBy);   // 32 MB (Q, pre-scaled by 1/64)
    __bf16* Kb   = (__bf16*)(w + 72 * MBy);   // 32 MB
    __bf16* Vtb  = (__bf16*)(w + 104 * MBy);  // 32 MB  Vt[B,H,DH,S]
    __bf16* Ctxb = (__bf16*)(w + 136 * MBy);  // 32 MB
    float*  AttnOut = (float*)(w + 40 * MBy); // 64 MB, reuses dead Q/K space

    const int nX = BB * SS * DD;   // 16,777,216
    const int nW = DD * DD;        //  1,048,576

    cvt_kernel<<<(nX + 255) / 256, 256, 0, stream>>>(enc, Xb, nX);
    cvt_kernel<<<(nW + 255) / 256, 256, 0, stream>>>(Wq, Wqb, nW);
    cvt_kernel<<<(nW + 255) / 256, 256, 0, stream>>>(Wk, Wkb, nW);
    cvt_kernel<<<(nW + 255) / 256, 256, 0, stream>>>(Wv, Wvb, nW);
    cvt_kernel<<<(nW + 255) / 256, 256, 0, stream>>>(Wo, Wob, nW);

    dim3 ggrid(BB * SS / 128, DD / 128);   // (128, 8)
    gemm_kernel<<<ggrid, 256, 0, stream>>>(Xb, Wqb, bq, Qb, 1.0f / 64.0f, 0);
    gemm_kernel<<<ggrid, 256, 0, stream>>>(Xb, Wkb, bk, Kb, 1.0f, 0);
    gemm_kernel<<<ggrid, 256, 0, stream>>>(Xb, Wvb, bv, Vtb, 1.0f, 1);

    attn_kernel<<<BB * HH * (SS / 128), 256, 0, stream>>>(Qb, Kb, Vtb, pad, Ctxb);

    gemm_kernel<<<ggrid, 256, 0, stream>>>(Ctxb, Wob, bo, AttnOut, 1.0f, 2);

    ln_kernel<<<BB * SS, 256, 0, stream>>>(enc, AttnOut, gamma, beta, (float*)d_out);
}
